// SelfAttention_56229711839264
// MI455X (gfx1250) — matrix operand order repaired and checked by reading, compiled, not run on hardware
//
#include <hip/hip_runtime.h>

// Problem constants (match reference: B=8, S=2048, D=1024, A=D/H=64)
#define BATCH   8
#define SEQ     2048
#define D_DIM   1024
#define A_DIM   64
#define M_TOTAL (BATCH * SEQ)          // 16384 rows total for projections
#define INV_SCALE 0.03125f             // 1/sqrt(1024)

typedef __attribute__((ext_vector_type(16))) __bf16 v16bf;
typedef __attribute__((ext_vector_type(8)))  __bf16 v8bf;
typedef __attribute__((ext_vector_type(8)))  float  v8f;

__device__ __forceinline__ v16bf cat8(v8bf lo, v8bf hi) {
    return __builtin_shufflevector(lo, hi, 0,1,2,3,4,5,6,7,8,9,10,11,12,13,14,15);
}

// A-matrix (16x32 bf16) element->K mapping (contiguous, matches B):
//   lanes 0-15: K 0..15 ; lanes 16-31: K 16..31
__device__ __forceinline__ int kkA(int e, int hi) {
    return e + (hi << 4);
}
// B-matrix (32x16 bf16) element->K mapping: lanes 0-15 K=0..15, lanes 16-31 K=16..31
// (contiguous in e -> two 16-byte vector loads when the source is K-contiguous)
__device__ __forceinline__ int kkB(int e, int hi) {
    return e + (hi << 4);
}

// ---------------------------------------------------------------------------
// Kernel 0: one-time weight pack.  Wt[64][1024](bf16) = transpose(W[1024][64])
// Moves the f32->bf16 conversion and the strided access out of the GEMM loop.
// grid = (256, 3), block = 256
// ---------------------------------------------------------------------------
__global__ __launch_bounds__(256) void pack_w_kernel(
    const float* __restrict__ Wq, const float* __restrict__ Wk, const float* __restrict__ Wv,
    unsigned short* __restrict__ Wtq, unsigned short* __restrict__ Wtk, unsigned short* __restrict__ Wtv)
{
    const float* W; unsigned short* O;
    switch (blockIdx.y) {
        case 0:  W = Wq; O = Wtq; break;
        case 1:  W = Wk; O = Wtk; break;
        default: W = Wv; O = Wtv; break;
    }
    const int idx = blockIdx.x * 256 + threadIdx.x;   // 0 .. 64*1024-1
    const int n = idx >> 10;          // output row    (0..63)
    const int k = idx & 1023;         // output column (0..1023)
    reinterpret_cast<__bf16*>(O)[idx] = (__bf16)W[k * A_DIM + n];
}

// ---------------------------------------------------------------------------
// Kernel 1: fused Q/K/V projection.  [16384,1024] @ [1024,64] + b -> bf16
// Q,K stored row-major [16384][64]; V stored TRANSPOSED [64][16384] so the
// P@V B-operand is K-contiguous later.
// grid = (M_TOTAL/16, 3), block = 128 (4 waves, one 16x16 tile per wave)
// ---------------------------------------------------------------------------
__global__ __launch_bounds__(128) void proj_kernel(
    const float* __restrict__ Xq, const float* __restrict__ Xk, const float* __restrict__ Xv,
    const unsigned short* __restrict__ Wtq, const unsigned short* __restrict__ Wtk,
    const unsigned short* __restrict__ Wtv,
    const float* __restrict__ bq, const float* __restrict__ bk, const float* __restrict__ bv,
    unsigned short* __restrict__ Yq, unsigned short* __restrict__ Yk, unsigned short* __restrict__ Yvt)
{
    const float* X; const unsigned short* Wt; const float* bias; unsigned short* Yraw;
    switch (blockIdx.y) {
        case 0:  X = Xq; Wt = Wtq; bias = bq; Yraw = Yq;  break;
        case 1:  X = Xk; Wt = Wtk; bias = bk; Yraw = Yk;  break;
        default: X = Xv; Wt = Wtv; bias = bv; Yraw = Yvt; break;
    }
    __bf16* Y = reinterpret_cast<__bf16*>(Yraw);
    const __bf16* Wtb = reinterpret_cast<const __bf16*>(Wt);

    const int wave = threadIdx.x >> 5;
    const int lane = threadIdx.x & 31;
    const int hi   = lane >> 4;
    const int lr   = lane & 15;
    const int m0   = blockIdx.x * 16;
    const int n0   = wave * 16;

    const float*  xrow  = X   + (size_t)(m0 + lr) * D_DIM;     // A row (f32)
    const __bf16* wtrow = Wtb + (size_t)(n0 + lr) * D_DIM;     // B column, K-contiguous (bf16)

    v8f acc = {};
    for (int k0 = 0; k0 < D_DIM; k0 += 32) {
        v16bf a;
        #pragma unroll
        for (int e = 0; e < 16; ++e)
            a[e] = (__bf16)xrow[k0 + kkA(e, hi)];              // contiguous runs of 8 -> b128
        const v16bf b = cat8(*(const v8bf*)(wtrow + k0 + (hi << 4)),
                             *(const v8bf*)(wtrow + k0 + (hi << 4) + 8));
        acc = __builtin_amdgcn_wmma_f32_16x16x32_bf16(false, a, false, b, (short)0, acc, false, false);
    }

    const int n = n0 + lr;
    const float bn = bias[n];
    if (blockIdx.y == 2) {
        // V transposed: Yvt[n][row], row = batch*SEQ + seqpos
        #pragma unroll
        for (int i = 0; i < 8; ++i)
            Y[(size_t)n * M_TOTAL + (m0 + i + hi * 8)] = (__bf16)(acc[i] + bn);
    } else {
        #pragma unroll
        for (int i = 0; i < 8; ++i)
            Y[(size_t)(m0 + i + hi * 8) * A_DIM + n] = (__bf16)(acc[i] + bn);
    }
}

// ---------------------------------------------------------------------------
// Kernel 2: fused attention.  One block = (batch, 16 query rows).
//   1) scores (16 x 2048) via WMMA into a 128 KB LDS tile (WGP has 320 KB)
//   2) softmax in LDS (float4 passes)
//   3) attn output tile streamed LDS -> HBM with global_store_async_from_lds_b128
//      (ASYNCcnt DMA path; overlaps with step 4 which only READS the LDS tile)
//   4) context tile (16 x 64) = P @ V via WMMA (Vt gives K-contiguous B loads)
// grid = (SEQ/16, BATCH), block = 128 (4 waves)
// ---------------------------------------------------------------------------
__global__ __launch_bounds__(128) void attn_kernel(
    const unsigned short* __restrict__ Qraw,
    const unsigned short* __restrict__ Kraw,
    const unsigned short* __restrict__ Vtraw,
    float* __restrict__ ctx, float* __restrict__ attn)
{
    extern __shared__ float smem[];
    float* P    = smem;                 // [16][SEQ] score/prob tile (128 KB, contiguous)
    float* redA = smem + 16 * SEQ;      // [128] row-max partials
    float* redB = redA + 128;           // [128] row-sum partials

    const __bf16* Qb  = reinterpret_cast<const __bf16*>(Qraw);
    const __bf16* Kb  = reinterpret_cast<const __bf16*>(Kraw);
    const __bf16* Vtb = reinterpret_cast<const __bf16*>(Vtraw);

    const int wave  = threadIdx.x >> 5;
    const int lane  = threadIdx.x & 31;
    const int hi    = lane >> 4;
    const int lr    = lane & 15;
    const int batch = blockIdx.y;
    const int m0    = blockIdx.x * 16;

    // --- q tile as two A fragments (K = 0..31, 32..63) ---
    const __bf16* qrow = Qb + (size_t)(batch * SEQ + m0 + lr) * A_DIM;
    const v16bf a0 = cat8(*(const v8bf*)(qrow +      hi * 16),
                          *(const v8bf*)(qrow +  8 + hi * 16));
    const v16bf a1 = cat8(*(const v8bf*)(qrow + 32 + hi * 16),
                          *(const v8bf*)(qrow + 40 + hi * 16));

    // --- scores: each wave handles every 4th 16-key tile (wave-uniform loop) ---
    const __bf16* kbase = Kb + (size_t)(batch * SEQ) * A_DIM;
    for (int nt = wave; nt < SEQ / 16; nt += 4) {
        const __bf16* krow = kbase + (size_t)(nt * 16 + lr) * A_DIM;
        const v16bf b0 = cat8(*(const v8bf*)(krow +      hi * 16),
                              *(const v8bf*)(krow +  8 + hi * 16));
        const v16bf b1 = cat8(*(const v8bf*)(krow + 32 + hi * 16),
                              *(const v8bf*)(krow + 40 + hi * 16));
        v8f s = {};
        s = __builtin_amdgcn_wmma_f32_16x16x32_bf16(false, a0, false, b0, (short)0, s, false, false);
        s = __builtin_amdgcn_wmma_f32_16x16x32_bf16(false, a1, false, b1, (short)0, s, false, false);
        #pragma unroll
        for (int i = 0; i < 8; ++i)
            P[(i + hi * 8) * SEQ + nt * 16 + lr] = s[i] * INV_SCALE;
    }
    __syncthreads();

    // --- softmax: 8 threads per row, 64 float4 (=256 floats) per thread ---
    const int row = threadIdx.x >> 3;
    const int sub = threadIdx.x & 7;
    float4* prow4 = (float4*)(P + row * SEQ + sub * 256);

    float mx = -3.4e38f;
    for (int j = 0; j < 64; ++j) {
        const float4 v = prow4[j];
        mx = fmaxf(mx, fmaxf(fmaxf(v.x, v.y), fmaxf(v.z, v.w)));
    }
    redA[row * 8 + sub] = mx;
    __syncthreads();

    float rm = redA[row * 8];
    #pragma unroll
    for (int i = 1; i < 8; ++i) rm = fmaxf(rm, redA[row * 8 + i]);

    float ls = 0.f;
    for (int j = 0; j < 64; ++j) {
        float4 v = prow4[j];
        v.x = __expf(v.x - rm); v.y = __expf(v.y - rm);
        v.z = __expf(v.z - rm); v.w = __expf(v.w - rm);
        prow4[j] = v;
        ls += (v.x + v.y) + (v.z + v.w);
    }
    redB[row * 8 + sub] = ls;
    __syncthreads();

    float rs = 0.f;
    #pragma unroll
    for (int i = 0; i < 8; ++i) rs += redB[row * 8 + i];
    const float inv = 1.f / rs;

    for (int j = 0; j < 64; ++j) {
        float4 v = prow4[j];
        v.x *= inv; v.y *= inv; v.z *= inv; v.w *= inv;
        prow4[j] = v;                       // final probs stay in LDS for P@V
    }
    __syncthreads();

    // --- stream the finished 128 KB attn tile LDS -> HBM via async DMA stores.
    // Tile rows m0..m0+15 are consecutive rows of attn[b], so the whole tile is
    // one contiguous 131072-byte global range; copy as 8192 16-byte chunks.
    {
        const unsigned ldsP = (unsigned)(uintptr_t)(void*)P;   // LDS offset (addr[31:0])
        float* gbase = attn + (size_t)(batch * SEQ + m0) * SEQ;
        for (int i = 0; i < 64; ++i) {                         // uniform loop, all lanes on
            const unsigned c    = (unsigned)threadIdx.x + 128u * (unsigned)i;
            const unsigned loff = ldsP + c * 16u;
            const unsigned long long gaddr =
                (unsigned long long)(uintptr_t)(gbase + (size_t)c * 4);
            asm volatile("global_store_async_from_lds_b128 %0, %1, off"
                         :: "v"(gaddr), "v"(loff) : "memory");
        }
    }

    // --- context tile: wave w owns output columns w*16..w*16+15, K loop over SEQ.
    // Overlaps with the in-flight async stores (both only read P). ---
    const int n0 = wave * 16;
    const __bf16* vtrow = Vtb + (size_t)(n0 + lr) * M_TOTAL + (size_t)batch * SEQ;
    v8f c = {};
    for (int ks = 0; ks < SEQ; ks += 32) {
        v16bf a;
        const float* pr = P + lr * SEQ + ks;
        #pragma unroll
        for (int p = 0; p < 8; ++p) {                          // kkA pairs are consecutive
            const float2 f2 = *(const float2*)(pr + kkA(2 * p, hi));
            a[2 * p]     = (__bf16)f2.x;
            a[2 * p + 1] = (__bf16)f2.y;
        }
        const v16bf b = cat8(*(const v8bf*)(vtrow + ks + (hi << 4)),
                             *(const v8bf*)(vtrow + ks + (hi << 4) + 8));
        c = __builtin_amdgcn_wmma_f32_16x16x32_bf16(false, a, false, b, (short)0, c, false, false);
    }
    #pragma unroll
    for (int i = 0; i < 8; ++i)
        ctx[(size_t)(batch * SEQ + m0 + i + hi * 8) * A_DIM + n0 + lr] = c[i];

    // Drain the async LDS->global stores before wave termination.
    asm volatile("s_wait_asynccnt 0x0" ::: "memory");
}

// ---------------------------------------------------------------------------
extern "C" void kernel_launch(void* const* d_in, const int* in_sizes, int n_in,
                              void* d_out, int out_size, void* d_ws, size_t ws_size,
                              hipStream_t stream) {
    (void)in_sizes; (void)n_in; (void)out_size; (void)ws_size;

    const float* query = (const float*)d_in[0];
    const float* key   = (const float*)d_in[1];
    const float* value = (const float*)d_in[2];
    const float* Wq    = (const float*)d_in[3];
    const float* bq    = (const float*)d_in[4];
    const float* Wk    = (const float*)d_in[5];
    const float* bk    = (const float*)d_in[6];
    const float* Wv    = (const float*)d_in[7];
    const float* bv    = (const float*)d_in[8];

    float* out  = (float*)d_out;
    float* ctx  = out;                                      // [B,S,A]
    float* attn = out + (size_t)BATCH * SEQ * A_DIM;        // [B,S,S]

    // Workspace: qb | kb | vt (2 MB each, bf16) then packed transposed weights.
    unsigned short* qb  = (unsigned short*)d_ws;
    unsigned short* kb  = qb  + (size_t)M_TOTAL * A_DIM;
    unsigned short* vt  = kb  + (size_t)M_TOTAL * A_DIM;    // [A_DIM][M_TOTAL]
    unsigned short* wtq = vt  + (size_t)M_TOTAL * A_DIM;
    unsigned short* wtk = wtq + (size_t)A_DIM * D_DIM;
    unsigned short* wtv = wtk + (size_t)A_DIM * D_DIM;

    pack_w_kernel<<<dim3((A_DIM * D_DIM) / 256, 3), 256, 0, stream>>>(
        Wq, Wk, Wv, wtq, wtk, wtv);

    proj_kernel<<<dim3(M_TOTAL / 16, 3), 128, 0, stream>>>(
        query, key, value, wtq, wtk, wtv, bq, bk, bv, qb, kb, vt);

    const size_t smem = (size_t)(16 * SEQ + 256) * sizeof(float);   // ~129 KB of 320 KB LDS
    attn_kernel<<<dim3(SEQ / 16, BATCH), 128, smem, stream>>>(
        qb, kb, vt, ctx, attn);
}
